// URHGNModel_41918880809420
// MI455X (gfx1250) — matrix operand order repaired and checked by reading, not compiled, not run on hardware
//
#include <hip/hip_runtime.h>

typedef __attribute__((ext_vector_type(16))) __bf16 v16bf;
typedef __attribute__((ext_vector_type(8)))  __bf16 v8bf;
typedef __attribute__((ext_vector_type(8)))  float  v8f;

#define NB 100000
#define NC 10000
#define EB 400000
#define EC 80000

static inline int ceil_div(int a, int b) { return (a + b - 1) / b; }

// ---------------- utility kernels ----------------

__global__ void k_fill_f32(float* __restrict__ p, float v, long long n) {
  long long i = (long long)blockIdx.x * blockDim.x + threadIdx.x;
  if (i < n) p[i] = v;
}

__global__ void k_cast_bf16(const float* __restrict__ s, __bf16* __restrict__ d, long long n) {
  long long i = (long long)blockIdx.x * blockDim.x + threadIdx.x;
  if (i < n) d[i] = (__bf16)s[i];
}

// Pack row-major W[K x N] (f32) into WMMA-B fragment order (bf16):
// block b = (k/32)*(N/16) + (n/16), offset = (n%16)*32 + (k%32).
// A lane (half,q) of subtile nt then reads its 16 fragment elements
// (K = half*16 + e) as 32 contiguous bytes at P[b*512 + q*32 + half*16].
__global__ void k_pack_b(const float* __restrict__ W, __bf16* __restrict__ P, int K, int N) {
  long long idx = (long long)blockIdx.x * blockDim.x + threadIdx.x;
  if (idx >= (long long)K * N) return;
  int k = (int)(idx / N), n = (int)(idx - (long long)k * N);
  size_t blk = (size_t)(k >> 5) * (N >> 4) + (n >> 4);
  P[blk * 512 + (size_t)(n & 15) * 32 + (k & 31)] = (__bf16)W[idx];
}

// ---------------- WMMA GEMM: C[f32, MxN] = A[bf16, MxK] * Bp[packed bf16] ----------------
// block = 256 threads = 8 waves. Wave w computes rows [((bx*8+w)*16) .. +16),
// cols [by*64 .. by*64+64). Packed B strip for the 64 columns is staged in LDS
// once per block (K <= 512 -> <= 64KB). Requires K % 32 == 0, N % 64 == 0.
__global__ void k_gemm_bf16(const __bf16* __restrict__ A, const __bf16* __restrict__ Bp,
                            float* __restrict__ C, int M, int N, int K) {
  __shared__ __bf16 ldsB[32768];   // 64KB: (K/32) blocks x 4 subtiles x 512
  const int wave = threadIdx.x >> 5;
  const int lane = threadIdx.x & 31;
  const int m0 = (blockIdx.x * 8 + wave) << 4;
  const int n0 = blockIdx.y << 6;
  const int nKb = K >> 5;

  // cooperative stage of packed B strip: per K-block, 4 consecutive subtile
  // blocks (2048 bf16 = 4KB) are contiguous in Bp.
  for (int kkb = 0; kkb < nKb; ++kkb) {
    const v8bf* src = (const v8bf*)(Bp + ((size_t)kkb * (N >> 4) + (n0 >> 4)) * 512);
    ((v8bf*)(ldsB + (size_t)kkb * 2048))[threadIdx.x] = src[threadIdx.x];
  }
  __syncthreads();

  if (m0 >= M) return;
  const int half = lane >> 4;   // 0: lanes 0-15, 1: lanes 16-31
  const int q = lane & 15;      // row (A) / column (B,C) within tile
  v8f acc0 = {}, acc1 = {}, acc2 = {}, acc3 = {};
  for (int kk = 0; kk < K; kk += 32) {
    // A fragment: lane q holds row m0+q; K(e) = half*16 + e
    const __bf16* Arow = A + (size_t)(m0 + q) * K + kk + half * 16;
    __builtin_prefetch(Arow + 32, 0, 1);
    const v8bf alo = *(const v8bf*)(Arow);
    const v8bf ahi = *(const v8bf*)(Arow + 8);
    v16bf a;
#pragma unroll
    for (int e = 0; e < 8; ++e) { a[e] = alo[e]; a[e + 8] = ahi[e]; }
    // B fragments from LDS: 32 contiguous bytes per lane per subtile
    const __bf16* Bl = ldsB + ((size_t)(kk >> 5) << 11) + q * 32 + half * 16;
    const v16bf b0 = *(const v16bf*)(Bl);
    const v16bf b1 = *(const v16bf*)(Bl + 512);
    const v16bf b2 = *(const v16bf*)(Bl + 1024);
    const v16bf b3 = *(const v16bf*)(Bl + 1536);
    acc0 = __builtin_amdgcn_wmma_f32_16x16x32_bf16(false, a, false, b0, (short)0, acc0, false, false);
    acc1 = __builtin_amdgcn_wmma_f32_16x16x32_bf16(false, a, false, b1, (short)0, acc1, false, false);
    acc2 = __builtin_amdgcn_wmma_f32_16x16x32_bf16(false, a, false, b2, (short)0, acc2, false, false);
    acc3 = __builtin_amdgcn_wmma_f32_16x16x32_bf16(false, a, false, b3, (short)0, acc3, false, false);
  }
  // C/D layout: VGPR r -> row half*8+r, lane q -> column
#pragma unroll
  for (int r = 0; r < 8; ++r) {
    float* Crow = C + (size_t)(m0 + half * 8 + r) * N + n0 + q;
    Crow[0]  = acc0[r];
    Crow[16] = acc1[r];
    Crow[32] = acc2[r];
    Crow[48] = acc3[r];
  }
}

// ---------------- GATv2 edge kernels ----------------

__device__ inline void atomicMaxF(float* addr, float val) {
  // int ordering == float ordering for non-negative; reversed unsigned for negative
  if (val >= 0.0f) atomicMax((int*)addr, __float_as_int(val));
  else             atomicMin((unsigned int*)addr, (unsigned int)__float_as_int(val));
}

// logits[e,h] = att[h] . leaky_relu(xl[src[e],h,:] + xr[dst[e],h,:]); also segment-max
__global__ void k_edge_logits(const float* __restrict__ xl, const float* __restrict__ xr,
                              const int* __restrict__ src, const int* __restrict__ dst,
                              int Ereal, int Etot, const float* __restrict__ att,
                              float* __restrict__ logits, float* __restrict__ mx,
                              int H, int D) {
  int idx = blockIdx.x * blockDim.x + threadIdx.x;
  if (idx >= Etot * H) return;
  int e = idx / H, h = idx - e * H;
  int s = (e < Ereal) ? src[e] : (e - Ereal);   // trailing self-loops
  int d = (e < Ereal) ? dst[e] : (e - Ereal);
  const float* pl = xl + ((size_t)s * H + h) * D;
  const float* pr = xr + ((size_t)d * H + h) * D;
  const float* pa = att + h * D;
  float acc = 0.0f;
  for (int k = 0; k < D; ++k) {
    float v = pl[k] + pr[k];
    v = (v > 0.0f) ? v : 0.2f * v;
    acc += v * pa[k];
  }
  logits[idx] = acc;
  atomicMaxF(&mx[(size_t)d * H + h], acc);
}

// p = exp(logit - max[dst]); segment-sum of p
__global__ void k_edge_exp(const int* __restrict__ dst, int Ereal, int Etot,
                           float* __restrict__ logits, const float* __restrict__ mx,
                           float* __restrict__ den, int H) {
  int idx = blockIdx.x * blockDim.x + threadIdx.x;
  if (idx >= Etot * H) return;
  int e = idx / H, h = idx - e * H;
  int d = (e < Ereal) ? dst[e] : (e - Ereal);
  float p = __expf(logits[idx] - mx[(size_t)d * H + h]);
  logits[idx] = p;
  atomicAdd(&den[(size_t)d * H + h], p);
}

// agg[dst,h,:] += (p/den[dst,h]) * xl[src,h,:]
__global__ void k_edge_agg(const float* __restrict__ p, const float* __restrict__ den,
                           const float* __restrict__ xl,
                           const int* __restrict__ src, const int* __restrict__ dst,
                           int Ereal, int Etot, float* __restrict__ agg, int H, int D) {
  int idx = blockIdx.x * blockDim.x + threadIdx.x;
  if (idx >= Etot * H) return;
  int e = idx / H, h = idx - e * H;
  int s = (e < Ereal) ? src[e] : (e - Ereal);
  int d = (e < Ereal) ? dst[e] : (e - Ereal);
  float alpha = p[idx] / den[(size_t)d * H + h];
  const float* pl = xl + ((size_t)s * H + h) * D;
  float* pd = agg + ((size_t)d * H + h) * D;
  for (int k = 0; k < D; ++k) atomicAdd(&pd[k], alpha * pl[k]);
}

// out_bf16 = relu(agg + bias)
__global__ void k_bias_relu_cast(const float* __restrict__ agg, const float* __restrict__ bias,
                                 __bf16* __restrict__ out, long long total, int N) {
  long long idx = (long long)blockIdx.x * blockDim.x + threadIdx.x;
  if (idx >= total) return;
  int j = (int)(idx % N);
  float v = agg[idx] + bias[j];
  out[idx] = (__bf16)(v > 0.0f ? v : 0.0f);
}

// combined[n, 0:64] = bf16(bfeat[n]); combined[n, 64:192] = cx2_bf16[map[n]]
__global__ void k_gather_concat(const float* __restrict__ bfeat, const __bf16* __restrict__ cx2,
                                const int* __restrict__ map, __bf16* __restrict__ comb, int M) {
  long long idx = (long long)blockIdx.x * blockDim.x + threadIdx.x;
  if (idx >= (long long)M * 192) return;
  int n = (int)(idx / 192), j = (int)(idx - (long long)n * 192);
  comb[idx] = (j < 64) ? (__bf16)bfeat[(size_t)n * 64 + j]
                       : cx2[(size_t)map[n] * 128 + (j - 64)];
}

// per-head fa MLP tail: relu(h+b1) @ w2[h] + b2[h] -> softmax(2) -> wsum += w
__global__ void k_fa_head(const float* __restrict__ hraw, const float* __restrict__ fa_b1,
                          const float* __restrict__ fa_w2, const float* __restrict__ fa_b2,
                          int head, float* __restrict__ wsum, int M) {
  int n = blockIdx.x * blockDim.x + threadIdx.x;
  if (n >= M) return;
  const float* b1 = fa_b1 + head * 128;
  const float* w2 = fa_w2 + head * 256;
  float l0 = fa_b2[head * 2], l1 = fa_b2[head * 2 + 1];
  const float* hr = hraw + (size_t)n * 128;
  for (int k = 0; k < 128; ++k) {
    float v = hr[k] + b1[k];
    v = v > 0.0f ? v : 0.0f;
    l0 += v * w2[k * 2];
    l1 += v * w2[k * 2 + 1];
  }
  float m = fmaxf(l0, l1);
  float e0 = __expf(l0 - m), e1 = __expf(l1 - m), s = e0 + e1;
  wsum[(size_t)n * 2]     += e0 / s;
  wsum[(size_t)n * 2 + 1] += e1 / s;
}

// fused (in place): scale building half by wmean0, community half by wmean1
__global__ void k_fuse_scale(__bf16* __restrict__ comb, const float* __restrict__ wsum, int M) {
  long long idx = (long long)blockIdx.x * blockDim.x + threadIdx.x;
  if (idx >= (long long)M * 192) return;
  int n = (int)(idx / 192), j = (int)(idx - (long long)n * 192);
  float w = (j < 64 ? wsum[(size_t)n * 2] : wsum[(size_t)n * 2 + 1]) * 0.125f;
  comb[idx] = (__bf16)((float)comb[idx] * w);
}

// tiny N=2 GEMM for b3: out[M,2] = X[M,K](bf16) @ W[K,2](f32)
__global__ void k_gemm_n2(const __bf16* __restrict__ X, const float* __restrict__ W,
                          int M, int K, float* __restrict__ out) {
  int n = blockIdx.x * blockDim.x + threadIdx.x;
  if (n >= M) return;
  float a0 = 0.0f, a1 = 0.0f;
  const __bf16* x = X + (size_t)n * K;
  for (int k = 0; k < K; ++k) {
    float v = (float)x[k];
    a0 += v * W[k * 2];
    a1 += v * W[k * 2 + 1];
  }
  out[(size_t)n * 2] = a0;
  out[(size_t)n * 2 + 1] = a1;
}

__global__ void k_logsoftmax2(const float* __restrict__ agg, const float* __restrict__ b,
                              float* __restrict__ out, int M) {
  int n = blockIdx.x * blockDim.x + threadIdx.x;
  if (n >= M) return;
  float v0 = agg[(size_t)n * 2] + b[0];
  float v1 = agg[(size_t)n * 2 + 1] + b[1];
  float m = fmaxf(v0, v1);
  float lse = m + logf(expf(v0 - m) + expf(v1 - m));
  out[(size_t)n * 2] = v0 - lse;
  out[(size_t)n * 2 + 1] = v1 - lse;
}

// ---------------- host orchestration ----------------

extern "C" void kernel_launch(void* const* d_in, const int* in_sizes, int n_in,
                              void* d_out, int out_size, void* d_ws, size_t ws_size,
                              hipStream_t stream) {
  const float* bfeat  = (const float*)d_in[0];
  const float* cfeat  = (const float*)d_in[1];
  const int*   be     = (const int*)d_in[2];
  const int*   ce     = (const int*)d_in[3];
  const int*   b2c    = (const int*)d_in[4];
  const float* c1_wl  = (const float*)d_in[5];
  const float* c1_wr  = (const float*)d_in[6];
  const float* c1_att = (const float*)d_in[7];
  const float* c1_b   = (const float*)d_in[8];
  const float* c2_wl  = (const float*)d_in[9];
  const float* c2_wr  = (const float*)d_in[10];
  const float* c2_att = (const float*)d_in[11];
  const float* c2_b   = (const float*)d_in[12];
  const float* fa_w1  = (const float*)d_in[13];
  const float* fa_b1  = (const float*)d_in[14];
  const float* fa_w2  = (const float*)d_in[15];
  const float* fa_b2  = (const float*)d_in[16];
  const float* b1_wl  = (const float*)d_in[17];
  const float* b1_wr  = (const float*)d_in[18];
  const float* b1_att = (const float*)d_in[19];
  const float* b1_b   = (const float*)d_in[20];
  const float* b2_wl  = (const float*)d_in[21];
  const float* b2_wr  = (const float*)d_in[22];
  const float* b2_att = (const float*)d_in[23];
  const float* b2_b   = (const float*)d_in[24];
  const float* b3_wl  = (const float*)d_in[25];
  const float* b3_wr  = (const float*)d_in[26];
  const float* b3_att = (const float*)d_in[27];
  const float* b3_b   = (const float*)d_in[28];

  const int* bsrc = be;      const int* bdst = be + EB;
  const int* csrc = ce;      const int* cdst = ce + EC;

  char* base = (char*)d_ws;
  size_t off = 0;
  auto alloc = [&](size_t bytes) -> void* {
    off = (off + 255) & ~(size_t)255;
    void* p = base + off;
    off += bytes;
    return p;
  };

  __bf16* w_c1l = (__bf16*)alloc((size_t)32 * 512 * 2);
  __bf16* w_c1r = (__bf16*)alloc((size_t)32 * 512 * 2);
  __bf16* w_c2l = (__bf16*)alloc((size_t)512 * 128 * 2);
  __bf16* w_c2r = (__bf16*)alloc((size_t)512 * 128 * 2);
  __bf16* w_fa1 = (__bf16*)alloc((size_t)8 * 192 * 128 * 2);
  __bf16* w_b1l = (__bf16*)alloc((size_t)192 * 512 * 2);
  __bf16* w_b1r = (__bf16*)alloc((size_t)192 * 512 * 2);
  __bf16* w_b2l = (__bf16*)alloc((size_t)512 * 128 * 2);
  __bf16* w_b2r = (__bf16*)alloc((size_t)512 * 128 * 2);
  __bf16* c_in  = (__bf16*)alloc((size_t)NC * 32 * 2);
  float*  c_xl  = (float*)alloc((size_t)NC * 512 * 4);
  float*  c_xr  = (float*)alloc((size_t)NC * 512 * 4);
  float*  c_log = (float*)alloc((size_t)(EC + NC) * 8 * 4);
  float*  c_mx  = (float*)alloc((size_t)NC * 8 * 4);
  float*  c_dn  = (float*)alloc((size_t)NC * 8 * 4);
  float*  c_agg = (float*)alloc((size_t)NC * 512 * 4);
  __bf16* c_a1  = (__bf16*)alloc((size_t)NC * 512 * 2);
  __bf16* c_a2  = (__bf16*)alloc((size_t)NC * 128 * 2);
  __bf16* comb  = (__bf16*)alloc((size_t)NB * 192 * 2);
  float*  fa_h  = (float*)alloc((size_t)NB * 128 * 4);
  float*  wsum  = (float*)alloc((size_t)NB * 2 * 4);
  float*  b_xl  = (float*)alloc((size_t)NB * 512 * 4);
  float*  b_xr  = (float*)alloc((size_t)NB * 512 * 4);
  float*  b_log = (float*)alloc((size_t)(EB + NB) * 8 * 4);
  float*  b_mx  = (float*)alloc((size_t)NB * 8 * 4);
  float*  b_dn  = (float*)alloc((size_t)NB * 8 * 4);
  float*  b_agg = (float*)alloc((size_t)NB * 512 * 4);
  __bf16* x1_bf = (__bf16*)alloc((size_t)NB * 512 * 2);
  __bf16* x2_bf = (__bf16*)alloc((size_t)NB * 128 * 2);
  float*  g3_xl = (float*)alloc((size_t)NB * 2 * 4);
  float*  g3_xr = (float*)alloc((size_t)NB * 2 * 4);
  float*  g3_ag = (float*)alloc((size_t)NB * 2 * 4);

  const int T = 256;
  auto g1 = [&](long long n) { return dim3((unsigned)((n + T - 1) / T)); };
  auto fill = [&](float* p, float v, long long n) {
    k_fill_f32<<<g1(n), T, 0, stream>>>(p, v, n);
  };
  auto cast = [&](const float* s, __bf16* d, long long n) {
    k_cast_bf16<<<g1(n), T, 0, stream>>>(s, d, n);
  };
  auto pack = [&](const float* W, __bf16* P, int K, int N) {
    k_pack_b<<<g1((long long)K * N), T, 0, stream>>>(W, P, K, N);
  };
  auto gemm = [&](const __bf16* A, const __bf16* Bp, float* C, int M, int N, int K) {
    dim3 grid((unsigned)ceil_div(ceil_div(M, 16), 8), (unsigned)(N / 64));
    k_gemm_bf16<<<grid, 256, 0, stream>>>(A, Bp, C, M, N, K);
  };
  auto gat = [&](float* xl, float* xr, const int* src, const int* dst, int Ereal, int n,
                 const float* att, float* logit, float* mx, float* dn, float* agg,
                 int H, int D) {
    int Etot = Ereal + n;  // self-loops appended
    fill(mx, -3.4e38f, (long long)n * H);
    fill(dn, 0.0f, (long long)n * H);
    fill(agg, 0.0f, (long long)n * H * D);
    long long tot = (long long)Etot * H;
    k_edge_logits<<<g1(tot), T, 0, stream>>>(xl, xr, src, dst, Ereal, Etot, att, logit, mx, H, D);
    k_edge_exp<<<g1(tot), T, 0, stream>>>(dst, Ereal, Etot, logit, mx, dn, H);
    k_edge_agg<<<g1(tot), T, 0, stream>>>(logit, dn, xl, src, dst, Ereal, Etot, agg, H, D);
  };

  // ---- pack weights to WMMA-B fragment order (bf16) ----
  pack(c1_wl, w_c1l, 32, 512);
  pack(c1_wr, w_c1r, 32, 512);
  pack(c2_wl, w_c2l, 512, 128);
  pack(c2_wr, w_c2r, 512, 128);
  for (int h = 0; h < 8; ++h)
    pack(fa_w1 + (size_t)h * 192 * 128, w_fa1 + (size_t)h * 192 * 128, 192, 128);
  pack(b1_wl, w_b1l, 192, 512);
  pack(b1_wr, w_b1r, 192, 512);
  pack(b2_wl, w_b2l, 512, 128);
  pack(b2_wr, w_b2r, 512, 128);
  cast(cfeat, c_in, (long long)NC * 32);

  // ---- community GATv2 layer 1 (32 -> 8x64) ----
  gemm(c_in, w_c1l, c_xl, NC, 512, 32);
  gemm(c_in, w_c1r, c_xr, NC, 512, 32);
  gat(c_xl, c_xr, csrc, cdst, EC, NC, c1_att, c_log, c_mx, c_dn, c_agg, 8, 64);
  k_bias_relu_cast<<<g1((long long)NC * 512), T, 0, stream>>>(c_agg, c1_b, c_a1, (long long)NC * 512, 512);

  // ---- community GATv2 layer 2 (512 -> 4x32) ----
  gemm(c_a1, w_c2l, c_xl, NC, 128, 512);
  gemm(c_a1, w_c2r, c_xr, NC, 128, 512);
  gat(c_xl, c_xr, csrc, cdst, EC, NC, c2_att, c_log, c_mx, c_dn, c_agg, 4, 32);
  k_bias_relu_cast<<<g1((long long)NC * 128), T, 0, stream>>>(c_agg, c2_b, c_a2, (long long)NC * 128, 128);

  // ---- gather community features, build combined [NB,192] ----
  k_gather_concat<<<g1((long long)NB * 192), T, 0, stream>>>(bfeat, c_a2, b2c, comb, NB);

  // ---- 8-head feature attention fusion ----
  fill(wsum, 0.0f, (long long)NB * 2);
  for (int h = 0; h < 8; ++h) {
    gemm(comb, w_fa1 + (size_t)h * 192 * 128, fa_h, NB, 128, 192);
    k_fa_head<<<g1(NB), T, 0, stream>>>(fa_h, fa_b1, fa_w2, fa_b2, h, wsum, NB);
  }
  k_fuse_scale<<<g1((long long)NB * 192), T, 0, stream>>>(comb, wsum, NB);

  // ---- building GATv2 layer 1 (192 -> 8x64) ----
  gemm(comb, w_b1l, b_xl, NB, 512, 192);
  gemm(comb, w_b1r, b_xr, NB, 512, 192);
  gat(b_xl, b_xr, bsrc, bdst, EB, NB, b1_att, b_log, b_mx, b_dn, b_agg, 8, 64);
  k_bias_relu_cast<<<g1((long long)NB * 512), T, 0, stream>>>(b_agg, b1_b, x1_bf, (long long)NB * 512, 512);

  // ---- building GATv2 layer 2 (512 -> 4x32) ----
  gemm(x1_bf, w_b2l, b_xl, NB, 128, 512);
  gemm(x1_bf, w_b2r, b_xr, NB, 128, 512);
  gat(b_xl, b_xr, bsrc, bdst, EB, NB, b2_att, b_log, b_mx, b_dn, b_agg, 4, 32);
  k_bias_relu_cast<<<g1((long long)NB * 128), T, 0, stream>>>(b_agg, b2_b, x2_bf, (long long)NB * 128, 128);

  // ---- building GATv2 layer 3 (128 -> 1x2) + log_softmax ----
  k_gemm_n2<<<g1(NB), T, 0, stream>>>(x2_bf, b3_wl, NB, 128, g3_xl);
  k_gemm_n2<<<g1(NB), T, 0, stream>>>(x2_bf, b3_wr, NB, 128, g3_xr);
  gat(g3_xl, g3_xr, bsrc, bdst, EB, NB, b3_att, b_log, b_mx, b_dn, g3_ag, 1, 2);
  k_logsoftmax2<<<g1(NB), T, 0, stream>>>(g3_ag, b3_b, (float*)d_out, NB);
}
